// SSA_35983236006158
// MI455X (gfx1250) — matrix operand order repaired and checked by reading, not compiled, not run on hardware
//
#include <hip/hip_runtime.h>

typedef unsigned short u16;
typedef unsigned int   u32;
typedef __attribute__((ext_vector_type(16))) __bf16 v16bf;
typedef __attribute__((ext_vector_type(8)))  float  v8f;

#define T_STEPS 32
#define NTOK    512
#define FDIM    512
#define NHEAD   8
#define HDIM    64
#define MROWS   (T_STEPS * NTOK)   /* 16384 rows in the (T*b, c) flattening */
#define BN_EPS  1e-5f

#define GBM 128
#define GBN 64
#define GBK 32
/* 32 + 8 pad: 80-byte rows keep every async-B128 LDS destination 16B-aligned
   AND make the A-fragment row reads bank-conflict-free (m*20 mod 64 distinct). */
#define GLS 40
/* ktv manual-transpose staging keeps the smaller pad (scalar u16 stores) */
#define TLS 34

union FragU { u32 u[8]; v16bf v; };

__device__ __forceinline__ u16 f2bf(float f){
  u32 u = __float_as_uint(f);
  u32 r = (u + 0x7fffu + ((u >> 16) & 1u)) >> 16;   // round-to-nearest-even
  return (u16)r;
}

__device__ __forceinline__ v8f vzero8(){
  v8f z;
#pragma unroll
  for (int i = 0; i < 8; ++i) z[i] = 0.0f;
  return z;
}

__device__ __forceinline__ v8f wmma_bf16(v16bf a, v16bf b, v8f c){
  // 8 args: (neg_a, A, neg_b, B, c_mod, C, reuse_a, reuse_b)
  return __builtin_amdgcn_wmma_f32_16x16x32_bf16(false, a, false, b, (short)0, c, false, false);
}

// Async copy: 16 bytes per lane, global -> LDS, tracked with ASYNCcnt.
// lds_byte is the 32-bit LDS byte address (low 32 bits of a generic pointer
// into a __shared__ array == the wave-relative LDS offset).
__device__ __forceinline__ void async_copy_b128(u32 lds_byte, const u16* g){
  asm volatile("global_load_async_to_lds_b128 %0, %1, off"
               :: "v"(lds_byte), "v"(g) : "memory");
}
__device__ __forceinline__ void wait_async_le3(){
  asm volatile("s_wait_asynccnt 0x3" ::: "memory");
}
__device__ __forceinline__ void wait_async_0(){
  asm volatile("s_wait_asynccnt 0x0" ::: "memory");
}

// A fragment, 16x32 bf16 (MxK), source row-major with row stride ldk.
// ISA layout: lane m = lane&15; VGPR i holds K pair k = 16*(i>>2) + 8*(lane>>4) + 2*(i&3).
__device__ __forceinline__ v16bf load_fragA(const u16* __restrict__ base, int ldk, int lane){
  FragU f;
  const u16* rp = base + (size_t)(lane & 15) * ldk + ((lane >> 4) << 3);
#pragma unroll
  for (int i = 0; i < 8; ++i){
    int k = ((i >> 2) << 4) + ((i & 3) << 1);
    f.u[i] = *(const u32*)(rp + k);
  }
  return f.v;
}

// B fragment, 32x16 bf16 (KxN), source stored TRANSPOSED as Bt[N][K] row-major so
// the per-lane K pairs are contiguous dwords: lane n = lane&15; VGPR i: k = 16*(i>>2) + 8*(lane>>4) + 2*(i&3).
__device__ __forceinline__ v16bf load_fragBt(const u16* __restrict__ base, int ldk, int lane){
  FragU f;
  const u16* rp = base + (size_t)(lane & 15) * ldk + ((lane >> 4) << 3);
#pragma unroll
  for (int i = 0; i < 8; ++i){
    f.u[i] = *(const u32*)(rp + (((i >> 2) << 4) + ((i & 3) << 1)));
  }
  return f.v;
}

// ---------------------------------------------------------------------------
// f32 -> bf16 conversion
// ---------------------------------------------------------------------------
__global__ __launch_bounds__(256)
void SSA_cvt(const float* __restrict__ in, u16* __restrict__ out, int n){
  int i = blockIdx.x * 256 + threadIdx.x;
  if (i < n) out[i] = f2bf(in[i]);
}

// ---------------------------------------------------------------------------
// C[M,N] (f32) = A[M,K] (bf16, row-major) @ B[N,K]^T (bf16, row-major) [+ bias]
// block 256 = 8 waves; macro tile 128x64; each wave a 32x32 patch (2x2 WMMA).
// K tiles staged via ASYNC global->LDS copies, double buffered:
//   per K-step each wave issues exactly 3 async B128 copies; async loads
//   complete in order, so s_wait_asynccnt<=3 guarantees the PREVIOUS tile
//   has landed while the next one is still in flight.
// ---------------------------------------------------------------------------
__global__ __launch_bounds__(256)
void SSA_gemm_bf16(const u16* __restrict__ A, const u16* __restrict__ B,
                   const float* __restrict__ bias, float* __restrict__ C,
                   int M, int N, int K)
{
  __shared__ __align__(16) u16 As[2][GBM * GLS];
  __shared__ __align__(16) u16 Bs[2][GBN * GLS];
  const int tid  = threadIdx.x;
  const int lane = tid & 31;
  const int wave = tid >> 5;
  const int wm = wave >> 1;         // 0..3
  const int wn = wave & 1;          // 0..1
  const int m0 = blockIdx.x * GBM;
  const int n0 = blockIdx.y * GBN;

  // cooperative staging: 128x32 A (16 bf16 = 2 x B128 per thread), 64x32 B (1 x B128)
  const int arow = tid >> 1;
  const int acol = (tid & 1) << 4;
  const int brow = tid >> 2;
  const int bcol = (tid & 3) << 3;
  const u16* gA = A + (size_t)(m0 + arow) * K + acol;
  const u16* gB = B + (size_t)(n0 + brow) * K + bcol;

  const u32 ldsA = (u32)(uintptr_t)&As[0][arow * GLS + acol];  // 16B aligned
  const u32 ldsB = (u32)(uintptr_t)&Bs[0][brow * GLS + bcol];  // 16B aligned
  const u32 bufA = (u32)(sizeof(u16) * GBM * GLS);             // 10240 B
  const u32 bufB = (u32)(sizeof(u16) * GBN * GLS);             //  5120 B

  v8f acc[2][2];
#pragma unroll
  for (int i = 0; i < 2; ++i)
#pragma unroll
    for (int j = 0; j < 2; ++j) acc[i][j] = vzero8();

  // prologue: tile 0 -> buffer 0
  async_copy_b128(ldsA,      gA);
  async_copy_b128(ldsA + 16, gA + 8);
  async_copy_b128(ldsB,      gB);

  int p = 0;
  for (int k0 = 0; k0 < K; k0 += GBK){
    const int nk = k0 + GBK;
    if (nk < K){                       // issue next tile into the other buffer
      const u32 q = (u32)(p ^ 1);
      async_copy_b128(ldsA + q * bufA,      gA + nk);
      async_copy_b128(ldsA + q * bufA + 16, gA + nk + 8);
      async_copy_b128(ldsB + q * bufB,      gB + nk);
      wait_async_le3();                // current tile's 3 copies have landed
    } else {
      wait_async_0();
    }
    __syncthreads();

    const u16* Ab = &As[p][0];
    const u16* Bb = &Bs[p][0];
    v16bf aF0 = load_fragA (Ab + (wm * 32 +  0) * GLS, GLS, lane);
    v16bf aF1 = load_fragA (Ab + (wm * 32 + 16) * GLS, GLS, lane);
    v16bf bF0 = load_fragBt(Bb + (wn * 32 +  0) * GLS, GLS, lane);
    v16bf bF1 = load_fragBt(Bb + (wn * 32 + 16) * GLS, GLS, lane);
    acc[0][0] = wmma_bf16(aF0, bF0, acc[0][0]);
    acc[0][1] = wmma_bf16(aF0, bF1, acc[0][1]);
    acc[1][0] = wmma_bf16(aF1, bF0, acc[1][0]);
    acc[1][1] = wmma_bf16(aF1, bF1, acc[1][1]);

    __syncthreads();                   // all waves done with buffer p before reuse
    p ^= 1;
  }

  // C layout: VGPR r -> M = r + 8*(lane>>4), N = lane&15
  const int cr = (lane >> 4) << 3;
  const int cc = lane & 15;
#pragma unroll
  for (int i = 0; i < 2; ++i){
#pragma unroll
    for (int j = 0; j < 2; ++j){
      int row = m0 + wm * 32 + i * 16 + cr;
      int col = n0 + wn * 32 + j * 16 + cc;
      float bv = bias ? bias[col] : 0.0f;
#pragma unroll
      for (int r = 0; r < 8; ++r)
        C[(size_t)(row + r) * N + col] = acc[i][j][r] + bv;
    }
  }
}

// ---------------------------------------------------------------------------
// BN statistics: per-channel sum / sumsq over the 16384 rows
// ---------------------------------------------------------------------------
__global__ __launch_bounds__(256)
void SSA_zero(float* __restrict__ p, int n){
  int i = blockIdx.x * 256 + threadIdx.x;
  if (i < n) p[i] = 0.0f;
}

__global__ __launch_bounds__(256)
void SSA_stats(const float* __restrict__ Y, float* __restrict__ stats){
  const int f = blockIdx.y * 256 + threadIdx.x;      // channel 0..511
  float s = 0.0f, s2 = 0.0f;
  for (int r = blockIdx.x; r < MROWS; r += 64){
    float xv = Y[(size_t)r * FDIM + f];
    s += xv; s2 += xv * xv;
  }
  atomicAdd(&stats[f], s);
  atomicAdd(&stats[FDIM + f], s2);
}

// ---------------------------------------------------------------------------
// (optional BN) + ParametricLIF scan over the 32 time steps.
// one thread per (token c, channel f); spikes written as bf16 (exact {0,1})
// and/or f32 to the final output buffer.
// ---------------------------------------------------------------------------
__global__ __launch_bounds__(256)
void SSA_bn_lif(const float* __restrict__ Y, const float* __restrict__ stats,
                const float* __restrict__ gamma, const float* __restrict__ beta,
                const float* __restrict__ vth_p, const float* __restrict__ tauw_p,
                u16* __restrict__ spk_out, float* __restrict__ f32_out)
{
  const int id = blockIdx.x * 256 + threadIdx.x;     // 0..262143
  const int f = id & (FDIM - 1);
  const int c = id >> 9;

  float sc = 1.0f, sh = 0.0f;
  if (gamma){
    float mean = stats[f] * (1.0f / (float)MROWS);
    float var  = stats[FDIM + f] * (1.0f / (float)MROWS) - mean * mean;
    float inv  = rsqrtf(var + BN_EPS);
    sc = gamma[f] * inv;
    sh = beta[f] - mean * sc;
  }
  const float vth   = vth_p[0];
  const float decay = 1.0f / (1.0f + __expf(-tauw_p[0]));   // sigmoid(tau_w)

  float v = 0.0f;
  for (int t = 0; t < T_STEPS; ++t){
    size_t idx = ((size_t)(t * NTOK + c)) * FDIM + f;
    float xv = Y[idx] * sc + sh;
    v += (xv - v) * decay;                 // charge
    float s = (v >= vth) ? 1.0f : 0.0f;    // fire
    v = (s > 0.0f) ? 0.0f : v;             // hard reset
    if (spk_out) spk_out[idx] = f2bf(s);
    if (f32_out) f32_out[idx] = s;
  }
}

// ---------------------------------------------------------------------------
// S^T = scale * (k^T v) per (time-step, head).  One workgroup per (tb, h).
// k/v tiles (32 tokens x 64 dims) staged transposed in LDS (manual u16
// scatter — async copies cannot transpose); 4x4 grid of 16x16 WMMA tiles,
// 2 tiles per wave.  S stored transposed (bf16) so the q@S kernel's B
// fragments read contiguous K pairs.
// ---------------------------------------------------------------------------
__global__ __launch_bounds__(256)
void SSA_ktv(const u16* __restrict__ Kspk, const u16* __restrict__ Vspk,
             u16* __restrict__ St)
{
  __shared__ u16 ktT[HDIM * TLS];
  __shared__ u16 vtT[HDIM * TLS];
  const int th = blockIdx.x;                 // tb*8 + h
  const int tb = th >> 3, h = th & 7;
  const int tid = threadIdx.x, lane = tid & 31, wave = tid >> 5;
  const int tm  = wave >> 1;                 // tile row (of 4)
  const int tn0 = (wave << 1) & 3;           // first tile col (of 4)

  v8f acc0 = vzero8(), acc1 = vzero8();

  const int ci = tid >> 3;                   // token within 32-tile
  const int db = (tid & 7) << 3;             // 8 dims per thread
  const size_t base = ((size_t)tb * NTOK) * FDIM + (size_t)h * HDIM;

  for (int c0 = 0; c0 < NTOK; c0 += 32){
    const size_t goff = base + (size_t)(c0 + ci) * FDIM + db;
    uint4 kv = *(const uint4*)(Kspk + goff);
    uint4 vv = *(const uint4*)(Vspk + goff);
    __syncthreads();
    const u16* kp = (const u16*)&kv;
    const u16* vp = (const u16*)&vv;
#pragma unroll
    for (int j = 0; j < 8; ++j){             // transpose into [dim][token]
      ktT[(db + j) * TLS + ci] = kp[j];
      vtT[(db + j) * TLS + ci] = vp[j];
    }
    __syncthreads();
    v16bf aF = load_fragA (ktT + (tm * 16) * TLS, TLS, lane);
    v16bf b0 = load_fragBt(vtT + (tn0 * 16) * TLS, TLS, lane);
    v16bf b1 = load_fragBt(vtT + (tn0 * 16 + 16) * TLS, TLS, lane);
    acc0 = wmma_bf16(aF, b0, acc0);
    acc1 = wmma_bf16(aF, b1, acc1);
  }

  const int cr = (lane >> 4) << 3;
  const int cc = lane & 15;
  u16* stb = St + (size_t)th * (HDIM * HDIM);
  const int m0 = tm * 16 + cr;
  const float scale = 0.125f;                // HEAD_DIM^-0.5 folded into S
#pragma unroll
  for (int r = 0; r < 8; ++r){
    stb[(size_t)(tn0 * 16 + cc) * HDIM + m0 + r]       = f2bf(acc0[r] * scale);
    stb[(size_t)((tn0 + 1) * 16 + cc) * HDIM + m0 + r] = f2bf(acc1[r] * scale);
  }
}

// ---------------------------------------------------------------------------
// W = q @ S  per (tb, h): M=512 tokens, N=64, K=64.  S is tiny (L2-resident),
// fragments read straight from global (q pairs contiguous, S^T pairs contiguous).
// ---------------------------------------------------------------------------
__global__ __launch_bounds__(256)
void SSA_qs(const u16* __restrict__ Qspk, const u16* __restrict__ St,
            float* __restrict__ W)
{
  const int ct = blockIdx.x, h = blockIdx.y, tb = blockIdx.z;
  const int tid = threadIdx.x, lane = tid & 31, wave = tid >> 5;
  const int wm = wave >> 1, wn = wave & 1;
  const int c0 = ct * 128 + wm * 32;
  const u16* qb = Qspk + ((size_t)tb * NTOK + c0) * FDIM + (size_t)h * HDIM;
  const u16* sb = St + ((size_t)(tb * NHEAD + h)) * (HDIM * HDIM);

  v8f acc[2][2];
#pragma unroll
  for (int i = 0; i < 2; ++i)
#pragma unroll
    for (int j = 0; j < 2; ++j) acc[i][j] = vzero8();

#pragma unroll
  for (int k0 = 0; k0 < HDIM; k0 += 32){
    v16bf a0 = load_fragA (qb + k0,             FDIM, lane);
    v16bf a1 = load_fragA (qb + 16 * FDIM + k0, FDIM, lane);
    v16bf b0 = load_fragBt(sb + (size_t)(wn * 32) * HDIM + k0,      HDIM, lane);
    v16bf b1 = load_fragBt(sb + (size_t)(wn * 32 + 16) * HDIM + k0, HDIM, lane);
    acc[0][0] = wmma_bf16(a0, b0, acc[0][0]);
    acc[0][1] = wmma_bf16(a0, b1, acc[0][1]);
    acc[1][0] = wmma_bf16(a1, b0, acc[1][0]);
    acc[1][1] = wmma_bf16(a1, b1, acc[1][1]);
  }

  const int cr = (lane >> 4) << 3;
  const int cc = lane & 15;
#pragma unroll
  for (int i = 0; i < 2; ++i){
#pragma unroll
    for (int j = 0; j < 2; ++j){
      int row = c0 + i * 16 + cr;
      int col = h * HDIM + wn * 32 + j * 16 + cc;
#pragma unroll
      for (int r = 0; r < 8; ++r)
        W[((size_t)tb * NTOK + row + r) * FDIM + col] = acc[i][j][r];
    }
  }
}

// ---------------------------------------------------------------------------
extern "C" void kernel_launch(void* const* d_in, const int* in_sizes, int n_in,
                              void* d_out, int out_size, void* d_ws, size_t ws_size,
                              hipStream_t stream)
{
  const float* x    = (const float*)d_in[0];
  const float* q_w  = (const float*)d_in[1];
  const float* k_w  = (const float*)d_in[2];
  const float* v_w  = (const float*)d_in[3];
  const float* p_w  = (const float*)d_in[4];
  const float* p_b  = (const float*)d_in[5];
  const float* q_g  = (const float*)d_in[6];  const float* q_be = (const float*)d_in[7];
  const float* k_g  = (const float*)d_in[8];  const float* k_be = (const float*)d_in[9];
  const float* v_g  = (const float*)d_in[10]; const float* v_be = (const float*)d_in[11];
  const float* p_g  = (const float*)d_in[12]; const float* p_be = (const float*)d_in[13];
  const float* q_vth = (const float*)d_in[14]; const float* q_tau = (const float*)d_in[15];
  const float* k_vth = (const float*)d_in[16]; const float* k_tau = (const float*)d_in[17];
  const float* v_vth = (const float*)d_in[18]; const float* v_tau = (const float*)d_in[19];
  const float* a_vth = (const float*)d_in[20]; const float* a_tau = (const float*)d_in[21];
  const float* pr_vth = (const float*)d_in[22]; const float* pr_tau = (const float*)d_in[23];

  char* wp = (char*)d_ws;
  auto carve = [&](size_t bytes) -> char* {
    char* r = wp; wp += (bytes + 255) & ~(size_t)255; return r;
  };
  u16*   xb    = (u16*)  carve((size_t)MROWS * FDIM * 2);
  u16*   wqb   = (u16*)  carve((size_t)FDIM * FDIM * 2);
  u16*   wkb   = (u16*)  carve((size_t)FDIM * FDIM * 2);
  u16*   wvb   = (u16*)  carve((size_t)FDIM * FDIM * 2);
  u16*   wpb   = (u16*)  carve((size_t)FDIM * FDIM * 2);
  float* y     = (float*)carve((size_t)MROWS * FDIM * 4);
  u16*   spkq  = (u16*)  carve((size_t)MROWS * FDIM * 2);
  u16*   spkk  = (u16*)  carve((size_t)MROWS * FDIM * 2);
  u16*   spkv  = (u16*)  carve((size_t)MROWS * FDIM * 2);
  u16*   stmat = (u16*)  carve((size_t)T_STEPS * NHEAD * HDIM * HDIM * 2);
  float* stats = (float*)carve((size_t)2 * FDIM * 4);
  u16*   spka  = spkq;   // q spikes are dead after SSA_qs -> reuse

  const int nx = MROWS * FDIM;
  const int nw = FDIM * FDIM;
  SSA_cvt<<<dim3((nx + 255) / 256), 256, 0, stream>>>(x,   xb,  nx);
  SSA_cvt<<<dim3((nw + 255) / 256), 256, 0, stream>>>(q_w, wqb, nw);
  SSA_cvt<<<dim3((nw + 255) / 256), 256, 0, stream>>>(k_w, wkb, nw);
  SSA_cvt<<<dim3((nw + 255) / 256), 256, 0, stream>>>(v_w, wvb, nw);
  SSA_cvt<<<dim3((nw + 255) / 256), 256, 0, stream>>>(p_w, wpb, nw);

  const dim3 ggrid(MROWS / GBM, FDIM / GBN);   // (128, 8)

  // --- q path ---
  SSA_gemm_bf16<<<ggrid, 256, 0, stream>>>(xb, wqb, nullptr, y, MROWS, FDIM, FDIM);
  SSA_zero <<<dim3(4), 256, 0, stream>>>(stats, 2 * FDIM);
  SSA_stats<<<dim3(64, 2), 256, 0, stream>>>(y, stats);
  SSA_bn_lif<<<dim3(1024), 256, 0, stream>>>(y, stats, q_g, q_be, q_vth, q_tau, spkq, nullptr);
  // --- k path ---
  SSA_gemm_bf16<<<ggrid, 256, 0, stream>>>(xb, wkb, nullptr, y, MROWS, FDIM, FDIM);
  SSA_zero <<<dim3(4), 256, 0, stream>>>(stats, 2 * FDIM);
  SSA_stats<<<dim3(64, 2), 256, 0, stream>>>(y, stats);
  SSA_bn_lif<<<dim3(1024), 256, 0, stream>>>(y, stats, k_g, k_be, k_vth, k_tau, spkk, nullptr);
  // --- v path ---
  SSA_gemm_bf16<<<ggrid, 256, 0, stream>>>(xb, wvb, nullptr, y, MROWS, FDIM, FDIM);
  SSA_zero <<<dim3(4), 256, 0, stream>>>(stats, 2 * FDIM);
  SSA_stats<<<dim3(64, 2), 256, 0, stream>>>(y, stats);
  SSA_bn_lif<<<dim3(1024), 256, 0, stream>>>(y, stats, v_g, v_be, v_vth, v_tau, spkv, nullptr);

  // --- attention: w = q @ (scale * k^T v) ---
  SSA_ktv<<<dim3(T_STEPS * NHEAD), 256, 0, stream>>>(spkk, spkv, stmat);
  SSA_qs <<<dim3(NTOK / 128, NHEAD, T_STEPS), 256, 0, stream>>>(spkq, stmat, y);
  SSA_bn_lif<<<dim3(1024), 256, 0, stream>>>(y, stats, nullptr, nullptr, a_vth, a_tau, spka, nullptr);

  // --- projection + BN + final LIF -> d_out (f32 spikes) ---
  SSA_gemm_bf16<<<ggrid, 256, 0, stream>>>(spka, wpb, p_b, y, MROWS, FDIM, FDIM);
  SSA_zero <<<dim3(4), 256, 0, stream>>>(stats, 2 * FDIM);
  SSA_stats<<<dim3(64, 2), 256, 0, stream>>>(y, stats);
  SSA_bn_lif<<<dim3(1024), 256, 0, stream>>>(y, stats, p_g, p_be, pr_vth, pr_tau, nullptr, (float*)d_out);

  (void)in_sizes; (void)n_in; (void)out_size; (void)ws_size;
}
